// Head_33062658244818
// MI455X (gfx1250) — matrix operand order repaired and checked by reading, not compiled, not run on hardware
//
#include <hip/hip_runtime.h>

#define B_ 16
#define S_ 2048
#define E_ 512
#define H_ 64

typedef __attribute__((ext_vector_type(16))) __bf16 v16bf;
typedef __attribute__((ext_vector_type(2)))  __bf16 v2bf;
typedef __attribute__((ext_vector_type(8)))  float  v8f;
typedef __attribute__((ext_vector_type(4)))  unsigned int v4u;
typedef __attribute__((ext_vector_type(4)))  int v4i;
typedef __attribute__((ext_vector_type(8)))  int v8i;

union FragAB {
  v16bf v;
  unsigned int u[8];
  uint4 q[2];
};

// Native bf16 conversion (lowers to v_cvt_pk_bf16_f32-class ops on gfx1250)
__device__ __forceinline__ unsigned short f2bf(float f) {
  __bf16 h = (__bf16)f;
  return __builtin_bit_cast(unsigned short, h);
}
__device__ __forceinline__ unsigned int pack2bf(float a, float b) {
  v2bf p;
  p.x = (__bf16)a;
  p.y = (__bf16)b;
  return __builtin_bit_cast(unsigned int, p);
}

__device__ __forceinline__ v8f wmma_bf16(const FragAB& a, const FragAB& b, v8f c) {
  return __builtin_amdgcn_wmma_f32_16x16x32_bf16(false, a.v, false, b.v,
                                                 (short)0, c, false, false);
}

// ---------------------------------------------------------------------------
// Phase 0: pre-swizzle Wq|Wk|Wv into B-fragment order, bf16-packed dwords.
// Layout: Wswz[((kc*12 + t)*32 + lane)*8 + r] where the fragment pair for
// lane (m = lane&15, g = lane>>4) is W[kc*32 + (r/4)*16 + g*8 + (r%4)*2 + {0,1}][ (t&3)*16+m ].
// Total: 16 kchunks * 12 tiles * 32 lanes * 8 dwords = 49152 dwords (192 KB).
// ---------------------------------------------------------------------------
__global__ __launch_bounds__(256) void wswz_kernel(
    const float* __restrict__ Wq, const float* __restrict__ Wk,
    const float* __restrict__ Wv, unsigned int* __restrict__ Wswz) {
  const int idx = blockIdx.x * 256 + threadIdx.x;
  if (idx >= 16 * 12 * 32 * 8) return;
  const int r    = idx & 7;
  const int lane = (idx >> 3) & 31;
  const int t    = (idx >> 8) % 12;
  const int kc   = (idx >> 8) / 12;
  const int g = lane >> 4, m = lane & 15;
  const int k   = kc * 32 + (r >> 2) * 16 + g * 8 + (r & 3) * 2;
  const int col = (t & 3) * 16 + m;
  const float* W = (t >> 2) == 0 ? Wq : ((t >> 2) == 1 ? Wk : Wv);
  Wswz[idx] = pack2bf(W[(size_t)k * H_ + col], W[(size_t)(k + 1) * H_ + col]);
}

// ---------------------------------------------------------------------------
// Phase 1: fused Q/K/V projection + padding mask + bf16 conversion.
// Each wave: one 16-row token tile, 12 WMMA accumulators (Q,K,V x 4 col tiles),
// K-loop over E=512 in steps of 32. A loads: b128 + packed cvt. B loads: b128
// from the pre-swizzled weight buffer. Q/K stored [B*S,64] bf16 (masked rows
// zeroed); V stored TRANSPOSED [B,64,S] bf16 (unmasked).
// ---------------------------------------------------------------------------
__global__ __launch_bounds__(128) void proj_kernel(
    const float* __restrict__ inp, const long long* __restrict__ lens,
    const unsigned int* __restrict__ Wswz,
    unsigned short* __restrict__ Qbf, unsigned short* __restrict__ Kbf,
    unsigned short* __restrict__ VbfT) {
  const int lane = threadIdx.x & 31;
  const int wv   = threadIdx.x >> 5;
  const int rt   = blockIdx.x * 4 + wv;     // row tile id, 0..2047
  const int m0   = rt * 16;                 // first global token row
  const int b    = m0 / S_;
  const int s0   = m0 % S_;
  const int m    = lane & 15;               // A: M index / B,C: N index
  const int g    = lane >> 4;               // lane group

  v8f acc[12];
#pragma unroll
  for (int t = 0; t < 12; ++t) acc[t] = (v8f)0.0f;

  const float4* arow4 = (const float4*)(inp + (size_t)(m0 + m) * E_);

  for (int kc = 0; kc < E_ / 32; ++kc) {
    FragAB a;  // A 16x32 bf16 fragment: pair k = kc*32 + (r/4)*16 + g*8 + (r%4)*2
#pragma unroll
    for (int h = 0; h < 2; ++h) {
      const int base4 = (kc * 32 + h * 16 + g * 8) >> 2;
      const float4 x0 = arow4[base4];
      const float4 x1 = arow4[base4 + 1];
      a.u[h * 4 + 0] = pack2bf(x0.x, x0.y);
      a.u[h * 4 + 1] = pack2bf(x0.z, x0.w);
      a.u[h * 4 + 2] = pack2bf(x1.x, x1.y);
      a.u[h * 4 + 3] = pack2bf(x1.z, x1.w);
    }
    const unsigned int* wbase = Wswz + ((size_t)kc * 12) * 256;
#pragma unroll
    for (int t = 0; t < 12; ++t) {
      const unsigned int* wf = wbase + (t * 32 + lane) * 8;
      FragAB bb;
      bb.q[0] = *(const uint4*)(wf);
      bb.q[1] = *(const uint4*)(wf + 4);
      acc[t] = wmma_bf16(a, bb, acc[t]);
    }
  }

  const int len = (int)lens[b];
  // C layout: row M = r + 8*g, col N = m
#pragma unroll
  for (int t = 0; t < 4; ++t) {
    const int col = t * 16 + m;
#pragma unroll
    for (int r = 0; r < 8; ++r) {
      const int srow = s0 + 8 * g + r;
      const bool pad = srow >= len;
      Qbf[(size_t)(b * S_ + srow) * H_ + col] = f2bf(pad ? 0.0f : acc[t][r]);
      Kbf[(size_t)(b * S_ + srow) * H_ + col] = f2bf(pad ? 0.0f : acc[4 + t][r]);
    }
  }
#pragma unroll
  for (int t = 0; t < 4; ++t) {
    const int col = t * 16 + m;
    unsigned int* vt = (unsigned int*)(VbfT + ((size_t)(b * H_ + col) * S_ + s0 + 8 * g));
#pragma unroll
    for (int r = 0; r < 4; ++r)
      vt[r] = pack2bf(acc[8 + t][2 * r], acc[8 + t][2 * r + 1]);
  }
}

// ---------------------------------------------------------------------------
// TDM: build a 2-D tensor descriptor (D#) per CDNA5 ISA and
// issue tensor_load_to_lds. data_size = 2 bytes (bf16).
// ---------------------------------------------------------------------------
__device__ __forceinline__ void tdm_load(unsigned int lds_off,
                                         unsigned long long gaddr,
                                         unsigned int tdim0, unsigned int tdim1,
                                         unsigned int tile0, unsigned int tile1,
                                         unsigned int stride0) {
  v4u g0;
  g0[0] = 1u;                                    // count=1, no gather
  g0[1] = lds_off;                               // lds_addr (bytes)
  g0[2] = (unsigned int)gaddr;                   // global_addr[31:0]
  g0[3] = (unsigned int)((gaddr >> 32) & 0x01FFFFFFull) | (2u << 30);  // type=2
  v8i g1;
  g1[0] = (int)(1u << 16);                       // data_size = 1 (2 bytes)
  g1[1] = (int)(tdim0 << 16);                    // tensor_dim0[15:0]
  g1[2] = (int)((tdim0 >> 16) | ((tdim1 & 0xFFFFu) << 16));
  g1[3] = (int)((tdim1 >> 16) | (tile0 << 16));  // tile_dim0
  g1[4] = (int)(tile1 & 0xFFFFu);                // tile_dim1 (tile_dim2=0)
  g1[5] = (int)stride0;                          // tensor_dim0_stride[31:0]
  g1[6] = 0;
  g1[7] = 0;
  v4i z4 = {0, 0, 0, 0};
#if __clang_major__ >= 23
  v8i z8 = {0, 0, 0, 0, 0, 0, 0, 0};
  __builtin_amdgcn_tensor_load_to_lds(g0, g1, z4, z4, z8, 0);
#else
  __builtin_amdgcn_tensor_load_to_lds(g0, g1, z4, z4, 0);
#endif
}

// ---------------------------------------------------------------------------
// Phase 2: fused flash-style attention. 4 waves/block, each wave owns a
// 16-query tile of one batch; the block streams K/V 32-key chunks through
// double-buffered LDS via the Tensor Data Mover. Exact softmax over all S
// (masked K rows are zero -> scores 0, matching the reference semantics).
// ---------------------------------------------------------------------------
__global__ __launch_bounds__(128) void attn_kernel(
    const unsigned short* __restrict__ Qbf, const unsigned short* __restrict__ Kbf,
    const unsigned short* __restrict__ VbfT, float* __restrict__ out) {
  __shared__ unsigned short Ksh[2][32 * 64];   // 32 keys x 64 dims
  __shared__ unsigned short Vsh[2][64 * 32];   // 64 dims x 32 keys (from V^T)
  __shared__ unsigned short Psh[4][16 * 32];   // per-wave P staging

  const int lane = threadIdx.x & 31;
  const int wv   = threadIdx.x >> 5;
  const int b    = blockIdx.x >> 5;             // 512 blocks: 16 batches x 32
  const int qg   = blockIdx.x & 31;
  const int q0   = (qg * 4 + wv) * 16;          // query row within batch
  const int m    = lane & 15;
  const int g    = lane >> 4;

  // Q as two A-fragments (dims 0..31 and 32..63), vectorized b128 loads
  FragAB qa0, qa1;
  const unsigned int* qrow = (const unsigned int*)(Qbf + (size_t)(b * S_ + q0 + m) * H_);
  qa0.q[0] = *(const uint4*)(qrow + g * 4);
  qa0.q[1] = *(const uint4*)(qrow + 8 + g * 4);
  qa1.q[0] = *(const uint4*)(qrow + 16 + g * 4);
  qa1.q[1] = *(const uint4*)(qrow + 24 + g * 4);

  v8f acc[4];
#pragma unroll
  for (int t = 0; t < 4; ++t) acc[t] = (v8f)0.0f;
  float mrun[8], lrun[8];
#pragma unroll
  for (int r = 0; r < 8; ++r) { mrun[r] = -1.0e30f; lrun[r] = 0.0f; }

  const unsigned long long kbase = (unsigned long long)(size_t)Kbf;
  const unsigned long long vbase = (unsigned long long)(size_t)VbfT;
  const int NCH = S_ / 32;  // 64 key chunks

  if (wv == 0) {
    tdm_load((unsigned int)(size_t)(void*)&Ksh[0][0],
             kbase + (unsigned long long)((size_t)(b * S_) * H_ * 2),
             H_, B_ * S_, H_, 32, H_);
    tdm_load((unsigned int)(size_t)(void*)&Vsh[0][0],
             vbase + (unsigned long long)((size_t)(b * H_) * S_ * 2),
             S_, B_ * H_, 32, H_, S_);
  }

  for (int i = 0; i < NCH; ++i) {
    const int buf = i & 1;
    if (wv == 0) {
      if (i + 1 < NCH) {
        tdm_load((unsigned int)(size_t)(void*)&Ksh[buf ^ 1][0],
                 kbase + (unsigned long long)((size_t)(b * S_ + (i + 1) * 32) * H_ * 2),
                 H_, B_ * S_, H_, 32, H_);
        tdm_load((unsigned int)(size_t)(void*)&Vsh[buf ^ 1][0],
                 vbase + (unsigned long long)(((size_t)(b * H_) * S_ + (i + 1) * 32) * 2),
                 S_, B_ * H_, 32, H_, S_);
        __builtin_amdgcn_s_wait_tensorcnt(2);  // two oldest (tile i) complete
      } else {
        __builtin_amdgcn_s_wait_tensorcnt(0);
      }
    }
    __syncthreads();  // tile i visible to all waves

    // ---- scores: two 16-key blocks, S = Q(16x64) . K^T ----
    v8f s[2];
#pragma unroll
    for (int kb = 0; kb < 2; ++kb) {
      const unsigned int* krow =
          (const unsigned int*)&Ksh[buf][(kb * 16 + m) * 64];  // key row = n
      FragAB b0, b1;
      b0.q[0] = *(const uint4*)(krow + g * 4);
      b0.q[1] = *(const uint4*)(krow + 8 + g * 4);
      b1.q[0] = *(const uint4*)(krow + 16 + g * 4);
      b1.q[1] = *(const uint4*)(krow + 24 + g * 4);
      v8f c = (v8f)0.0f;
      c = wmma_bf16(qa0, b0, c);
      c = wmma_bf16(qa1, b1, c);
      s[kb] = c * 0.125f;  // 1/sqrt(64)
    }

    // ---- online softmax: per-row max via width-16 butterfly ----
    float rm[8];
#pragma unroll
    for (int r = 0; r < 8; ++r) rm[r] = fmaxf(s[0][r], s[1][r]);
#pragma unroll
    for (int d = 1; d < 16; d <<= 1)
#pragma unroll
      for (int r = 0; r < 8; ++r) rm[r] = fmaxf(rm[r], __shfl_xor(rm[r], d, 32));

    float p0[8], p1[8], sc[8];
#pragma unroll
    for (int r = 0; r < 8; ++r) {
      const float mn = fmaxf(mrun[r], rm[r]);
      sc[r]   = __expf(mrun[r] - mn);
      mrun[r] = mn;
      p0[r]   = __expf(s[0][r] - mn);
      p1[r]   = __expf(s[1][r] - mn);
      lrun[r] = lrun[r] * sc[r] + p0[r] + p1[r];
    }
#pragma unroll
    for (int t = 0; t < 4; ++t)
#pragma unroll
      for (int r = 0; r < 8; ++r) acc[t][r] = acc[t][r] * sc[r];

    // ---- C-layout P -> A-fragment layout via LDS bounce ----
#pragma unroll
    for (int r = 0; r < 8; ++r) {
      const int row = r + 8 * g;
      Psh[wv][row * 32 + m]      = f2bf(p0[r]);
      Psh[wv][row * 32 + 16 + m] = f2bf(p1[r]);
    }
    asm volatile("s_wait_dscnt 0x0" ::: "memory");
    FragAB pa;
    const unsigned int* prow = (const unsigned int*)&Psh[wv][m * 32];
    pa.q[0] = *(const uint4*)(prow + g * 4);
    pa.q[1] = *(const uint4*)(prow + 8 + g * 4);

    // ---- out += P(16x32) . V(32x64), 4 N-tiles ----
#pragma unroll
    for (int t = 0; t < 4; ++t) {
      const unsigned int* vrow =
          (const unsigned int*)&Vsh[buf][(t * 16 + m) * 32];  // dim row = col n
      FragAB vb;
      vb.q[0] = *(const uint4*)(vrow + g * 4);
      vb.q[1] = *(const uint4*)(vrow + 8 + g * 4);
      acc[t] = wmma_bf16(pa, vb, acc[t]);
    }
    __syncthreads();  // everyone done with buf before it is refilled
  }

  // ---- finalize: row sum of l across the 16-lane group, normalize, store ----
  float linv[8];
#pragma unroll
  for (int r = 0; r < 8; ++r) {
    float l = lrun[r];
#pragma unroll
    for (int d = 1; d < 16; d <<= 1) l += __shfl_xor(l, d, 32);
    linv[r] = l > 0.0f ? 1.0f / l : 0.0f;  // guards the NaN->0 path
  }
#pragma unroll
  for (int t = 0; t < 4; ++t)
#pragma unroll
    for (int r = 0; r < 8; ++r)
      out[(size_t)(b * S_ + q0 + r + 8 * g) * H_ + t * 16 + m] = acc[t][r] * linv[r];
}

// ---------------------------------------------------------------------------
extern "C" void kernel_launch(void* const* d_in, const int* in_sizes, int n_in,
                              void* d_out, int out_size, void* d_ws, size_t ws_size,
                              hipStream_t stream) {
  const float* inp      = (const float*)d_in[0];
  const long long* lens = (const long long*)d_in[1];
  const float* Wq       = (const float*)d_in[2];
  const float* Wk       = (const float*)d_in[3];
  const float* Wv       = (const float*)d_in[4];
  float* out            = (float*)d_out;

  unsigned short* Qbf  = (unsigned short*)d_ws;                 // 4 MB
  unsigned short* Kbf  = Qbf + (size_t)B_ * S_ * H_;            // 4 MB
  unsigned short* VbfT = Kbf + (size_t)B_ * S_ * H_;            // 4 MB ([B,64,S])
  unsigned int*   Wswz = (unsigned int*)(VbfT + (size_t)B_ * H_ * S_);  // 192 KB

  wswz_kernel<<<192, 256, 0, stream>>>(Wq, Wk, Wv, Wswz);
  proj_kernel<<<512, 128, 0, stream>>>(inp, lens, Wswz, Qbf, Kbf, VbfT);
  attn_kernel<<<512, 128, 0, stream>>>(Qbf, Kbf, VbfT, out);
}
